// FocalAtt_3478923510578
// MI455X (gfx1250) — compile-verified, matrix operand order checked by reading, not run on hardware
//
#include <hip/hip_runtime.h>
#include <hip/hip_bf16.h>
#include <math.h>

// ===== CDNA5 (gfx1250) WMMA + TDM implementation of FocalAtt cross-attention =====
// GEMMs: v_wmma_f32_16x16x32_bf16; operand staging: tensor_load_to_lds (TDM)
// with double-buffered LDS and s_wait_tensorcnt pipelining. wave32, 8 waves/block.

typedef __bf16 bf16_t;
typedef __attribute__((ext_vector_type(16))) __bf16 v16bf;
typedef __attribute__((ext_vector_type(8)))  __bf16 v8bf;
typedef __attribute__((ext_vector_type(8)))  float  v8f;
typedef __attribute__((ext_vector_type(4)))  unsigned int u32x4;
typedef __attribute__((ext_vector_type(8)))  unsigned int u32x8;

#define B_SZ 8
#define LQ_  512
#define LK_  1024
#define QD_  768
#define HID_ 1024
#define MID_ 4096
#define NH_  16
#define DH_  64

#define USE_TDM 1   // 0 -> manual cooperative global->LDS loads

// ---------------------------------------------------------------------------
// TDM: issue a 2D bf16 tile load (tile_d0 x tile_d1, row stride = stride_elems)
// into LDS at byte offset lds_off.  LDS rows padded 64B->80B (pad_interval=16DW,
// pad_amount=4DW) to match LDSP=40 elements.  D# layout per CDNA5 ISA ch.8.
// ---------------------------------------------------------------------------
__device__ __forceinline__ void tdm_load_2d(unsigned lds_off, const void* gptr,
                                            int tile_d0, int tile_d1,
                                            int stride_elems) {
  unsigned long long ga = (unsigned long long)gptr;
  u32x4 g0;
  g0[0] = 1u;                                   // count=1, user mode
  g0[1] = lds_off;                              // lds_addr (bytes)
  g0[2] = (unsigned)ga;                         // global_addr[31:0]
  g0[3] = (unsigned)((ga >> 32) & 0x1FFFFFFu)   // global_addr[56:32]
        | (2u << 30);                           // type=2 ("image")
  u32x8 g1;
  g1[0] = (1u << 16)                            // data_size = 2 bytes
        | (1u << 20)                            // pad_enable
        | (3u << 22)                            // pad_interval: 16 DWORDs
        | (3u << 25);                           // pad_amount : 4 DWORDs
  g1[1] = ((unsigned)tile_d0 & 0xFFFFu) << 16;  // tensor_dim0[15:0]
  g1[2] = (((unsigned)tile_d0 >> 16) & 0xFFFFu)          // tensor_dim0[31:16]
        | (((unsigned)tile_d1 & 0xFFFFu) << 16);         // tensor_dim1[15:0]
  g1[3] = (((unsigned)tile_d1 >> 16) & 0xFFFFu)          // tensor_dim1[31:16]
        | (((unsigned)tile_d0 & 0xFFFFu) << 16);         // tile_dim0
  g1[4] = ((unsigned)tile_d1 & 0xFFFFu);                 // tile_dim1 (tile_dim2=0)
  g1[5] = (unsigned)stride_elems;                        // tensor_dim0_stride lo32
  g1[6] = 0u; g1[7] = 0u;
  u32x4 gz = {0u, 0u, 0u, 0u};                  // groups 2/3 unused (2D tensor)
  asm volatile("tensor_load_to_lds %0, %1, %2, %3"
               :: "s"(g0), "s"(g1), "s"(gz), "s"(gz)
               : "memory");
}

// ---------------------------------------------------------------------------
// Generic batched GEMM:  C[z] = A[z] (MxK, bf16, row-major) * B[z] (KxN),
// B supplied transposed (Bt: NxK row-major).  Epilogue as in round 0.
// ---------------------------------------------------------------------------
struct GemmP {
  const bf16_t* A;   long long aB;  int lda;
  const bf16_t* Bt;  long long bB;  int ldbt;
  const float*  bias;
  float scale; int relu; int bdiv;
  float*  outF; long long f_s1, f_s2, f_sM, f_sH, f_sD;
  bf16_t* outB; long long o_s1, o_s2, o_sM, o_sH, o_sD;
  const bf16_t* v0; float* xacc;   // bilinear fuse (z = b*16+h)
  int K;
};

#define BM 128
#define BK 32
#define LDSP 40   // padded LDS row stride in elements (80B, keeps 16B align)

template <int BNT>
__global__ __launch_bounds__(256) void gemm_wmma(GemmP p) {
  constexpr int TN = BNT / 32;   // 16-col tiles per wave (2 or 4)
  __shared__ __align__(16) bf16_t Asl[2][BM  * LDSP];
  __shared__ __align__(16) bf16_t Bsl[2][BNT * LDSP];

  const int tid  = threadIdx.x;
  const int lane = tid & 31;
  const int wave = tid >> 5;
  const int wr   = wave >> 1;            // 0..3 : 32-row strip
  const int wc   = wave & 1;             // 0..1 : (TN*16)-col strip
  const int z    = blockIdx.z;
  const int m0   = blockIdx.y * BM;
  const int n0   = blockIdx.x * BNT;

  const bf16_t* Ab  = p.A  + (long long)z * p.aB + (long long)m0 * p.lda;
  const bf16_t* Btb = p.Bt + (long long)z * p.bB + (long long)n0 * p.ldbt;

  v8f acc[2][TN] = {};

  const int frow = lane & 15;            // fragment row (ISA 7.12.2)
  const int fkb  = (lane >> 4) << 3;     // K sub-block: 0 or 8

  const int nk = p.K / BK;

#if USE_TDM
  const unsigned ldsA0 = (unsigned)(unsigned long long)&Asl[0][0];
  const unsigned ldsA1 = (unsigned)(unsigned long long)&Asl[1][0];
  const unsigned ldsB0 = (unsigned)(unsigned long long)&Bsl[0][0];
  const unsigned ldsB1 = (unsigned)(unsigned long long)&Bsl[1][0];
  if (wave == 0) tdm_load_2d(ldsA0, Ab,  BK, BM,  p.lda);
  if (wave == 1) tdm_load_2d(ldsB0, Btb, BK, BNT, p.ldbt);
#endif

  for (int kk = 0; kk < nk; ++kk) {
    const int cur = kk & 1;
#if USE_TDM
    if (kk + 1 < nk) {   // prefetch next tile into other buffer
      if (wave == 0) tdm_load_2d(cur ? ldsA0 : ldsA1, Ab  + (long long)(kk + 1) * BK, BK, BM,  p.lda);
      if (wave == 1) tdm_load_2d(cur ? ldsB0 : ldsB1, Btb + (long long)(kk + 1) * BK, BK, BNT, p.ldbt);
      if (wave < 2) __builtin_amdgcn_s_wait_tensorcnt(1);
    } else {
      if (wave < 2) __builtin_amdgcn_s_wait_tensorcnt(0);
    }
    __syncthreads();
#else
    { // manual cooperative load (fallback)
      const int k0 = kk * BK;
      for (int vi = tid; vi < (BM * BK) / 8; vi += 256) {
        int r = vi >> 2, c = (vi & 3) << 3;
        *(v8bf*)&Asl[cur][r * LDSP + c] = *(const v8bf*)&Ab[(long long)r * p.lda + k0 + c];
      }
      for (int vi = tid; vi < (BNT * BK) / 8; vi += 256) {
        int r = vi >> 2, c = (vi & 3) << 3;
        *(v8bf*)&Bsl[cur][r * LDSP + c] = *(const v8bf*)&Btb[(long long)r * p.ldbt + k0 + c];
      }
      __syncthreads();
    }
#endif

    v16bf afr[2], bfr[TN];
#pragma unroll
    for (int t = 0; t < 2; ++t) {
      int ar = wr * 32 + t * 16 + frow;
      v8bf alo = *(const v8bf*)&Asl[cur][ar * LDSP + fkb];
      v8bf ahi = *(const v8bf*)&Asl[cur][ar * LDSP + fkb + 16];
      afr[t] = __builtin_shufflevector(alo, ahi, 0,1,2,3,4,5,6,7,8,9,10,11,12,13,14,15);
    }
#pragma unroll
    for (int t = 0; t < TN; ++t) {
      int br = wc * (TN * 16) + t * 16 + frow;
      v8bf blo = *(const v8bf*)&Bsl[cur][br * LDSP + fkb];
      v8bf bhi = *(const v8bf*)&Bsl[cur][br * LDSP + fkb + 16];
      bfr[t] = __builtin_shufflevector(blo, bhi, 0,1,2,3,4,5,6,7,8,9,10,11,12,13,14,15);
    }
#pragma unroll
    for (int i = 0; i < 2; ++i)
#pragma unroll
      for (int j = 0; j < TN; ++j)
        acc[i][j] = __builtin_amdgcn_wmma_f32_16x16x32_bf16(
            false, afr[i], false, bfr[j], (short)0, acc[i][j], false, false);
    __syncthreads();
  }

  // ----- epilogue -----
  const int cn = lane & 15;
  const int mh = (lane >> 4) << 3;

  if (p.v0) {  // bilinear fuse: x[b, h*64+n] += sum_m acc*v0   (z = b*16+h)
#pragma unroll
    for (int j = 0; j < TN; ++j) {
      int n = n0 + wc * (TN * 16) + j * 16 + cn;
      float s = 0.f;
#pragma unroll
      for (int i = 0; i < 2; ++i)
#pragma unroll
        for (int r = 0; r < 8; ++r) {
          int m = m0 + wr * 32 + i * 16 + mh + r;
          s += acc[i][j][r] *
               (float)p.v0[(long long)z * (LK_ * DH_) + (long long)m * DH_ + n];
        }
      atomicAdd(&p.xacc[(long long)(z >> 4) * HID_ + (long long)(z & 15) * DH_ + n], s);
    }
    return;
  }

  long long zf = (long long)(z / p.bdiv) * p.f_s1 + (long long)(z % p.bdiv) * p.f_s2;
  long long zo = (long long)(z / p.bdiv) * p.o_s1 + (long long)(z % p.bdiv) * p.o_s2;
#pragma unroll
  for (int i = 0; i < 2; ++i)
#pragma unroll
    for (int j = 0; j < TN; ++j)
#pragma unroll
      for (int r = 0; r < 8; ++r) {
        int m = m0 + wr * 32 + i * 16 + mh + r;
        int n = n0 + wc * (TN * 16) + j * 16 + cn;
        float v = acc[i][j][r] * p.scale;
        if (p.bias) v += p.bias[n];
        if (p.relu) v = v > 0.f ? v : 0.f;
        int h = n >> 6, d = n & 63;
        if (p.outF)
          p.outF[zf + (long long)m * p.f_sM + (long long)h * p.f_sH + (long long)d * p.f_sD] = v;
        if (p.outB)
          p.outB[zo + (long long)m * p.o_sM + (long long)h * p.o_sH + (long long)d * p.o_sD] = (bf16_t)v;
      }
}

// ---------------------------------------------------------------------------
// softmax over last dim (Lk=1024); writes att (bf16) and attT (bf16)
// ---------------------------------------------------------------------------
__global__ __launch_bounds__(256) void softmax_kernel(const float* scores,
                                                      bf16_t* att, bf16_t* attT) {
  const int l = blockIdx.x;
  const long long z = blockIdx.y;
  const int tid = threadIdx.x;
  const float* row = scores + (z * LQ_ + l) * (long long)LK_;
  __shared__ float red[256];

  float v[4]; float mx = -3.4e38f;
#pragma unroll
  for (int i = 0; i < 4; ++i) { v[i] = row[tid + i * 256]; mx = fmaxf(mx, v[i]); }
  red[tid] = mx; __syncthreads();
  for (int o = 128; o > 0; o >>= 1) { if (tid < o) red[tid] = fmaxf(red[tid], red[tid + o]); __syncthreads(); }
  mx = red[0]; __syncthreads();

  float s = 0.f;
#pragma unroll
  for (int i = 0; i < 4; ++i) { v[i] = __expf(v[i] - mx); s += v[i]; }
  red[tid] = s; __syncthreads();
  for (int o = 128; o > 0; o >>= 1) { if (tid < o) red[tid] += red[tid + o]; __syncthreads(); }
  float inv = 1.f / red[0];

#pragma unroll
  for (int i = 0; i < 4; ++i) {
    int c = tid + i * 256;
    bf16_t a = (bf16_t)(v[i] * inv);
    att [(z * LQ_ + l) * (long long)LK_ + c] = a;
    attT[(z * LK_ + c) * (long long)LQ_ + l] = a;
  }
}

// att_out[b,l,c] = mean_h relu(scores[b,h,l,c])
__global__ void attout_kernel(const float* scores, float* out) {
  const int l = blockIdx.x, b = blockIdx.y, tid = threadIdx.x;
  for (int c = tid; c < LK_; c += 256) {
    float s = 0.f;
    for (int h = 0; h < NH_; ++h) {
      const float* pp = &scores[(((long long)(b * NH_ + h) * LQ_ + l) * LK_) + c];
      if (h + 1 < NH_)
        __builtin_prefetch(pp + (long long)LQ_ * LK_, 0, 0);  // global_prefetch_b8
      float x = *pp;
      s += x > 0.f ? x : 0.f;
    }
    out[((long long)b * LQ_ + l) * LK_ + c] = s * (1.f / NH_);
  }
}

// torch-style LN over n=1024: a*(x-mean)/(sqrt(var_unbiased)+1e-6)+b
__global__ __launch_bounds__(256) void ln_kernel(const float* in1, const float* in2,
                                                 const float* ga, const float* be,
                                                 float* outF, bf16_t* outB, int n) {
  const long long row = blockIdx.x;
  const int tid = threadIdx.x;
  const float* p1 = in1 + row * n;
  const float* p2 = in2 ? in2 + row * n : nullptr;
  __shared__ float red[256];

  float v[4]; float s = 0.f;
#pragma unroll
  for (int i = 0; i < 4; ++i) {
    int c = tid + i * 256;
    v[i] = p1[c] + (p2 ? p2[c] : 0.f);
    s += v[i];
  }
  red[tid] = s; __syncthreads();
  for (int o = 128; o > 0; o >>= 1) { if (tid < o) red[tid] += red[tid + o]; __syncthreads(); }
  float mean = red[0] / (float)n; __syncthreads();

  s = 0.f;
#pragma unroll
  for (int i = 0; i < 4; ++i) { float d = v[i] - mean; s += d * d; }
  red[tid] = s; __syncthreads();
  for (int o = 128; o > 0; o >>= 1) { if (tid < o) red[tid] += red[tid + o]; __syncthreads(); }
  float var = red[0] / (float)(n - 1);
  float inv = 1.f / (sqrtf(var) + 1e-6f);

#pragma unroll
  for (int i = 0; i < 4; ++i) {
    int c = tid + i * 256;
    float y = ga[c] * (v[i] - mean) * inv + be[c];
    if (outF) outF[row * n + c] = y;
    if (outB) outB[row * n + c] = (bf16_t)y;
  }
}

// x + positional encoding; write bf16 (and optionally f32)
__global__ void addpe_kernel(const float* x, bf16_t* xb, float* xf, int L, int D) {
  long long idx = (long long)blockIdx.x * 256 + threadIdx.x;
  int c = (int)(idx % D);
  int l = (int)((idx / D) % L);
  float i2  = (float)(c & ~1);
  float ang = (float)l * __expf(i2 * (-9.210340371976184f / (float)D));
  float pe  = (c & 1) ? __cosf(ang) : __sinf(ang);
  float v = x[idx] + pe;
  xb[idx] = (bf16_t)v;
  if (xf) xf[idx] = v;
}

__global__ void cvt_kernel(const float* x, bf16_t* y) {
  long long i = (long long)blockIdx.x * 256 + threadIdx.x;
  y[i] = (bf16_t)x[i];
}

__global__ void zero_kernel(float* p, int n) {
  int i = blockIdx.x * 256 + threadIdx.x;
  if (i < n) p[i] = 0.f;
}

// W [K][N] f32 -> Wt [N][K] bf16
__global__ void transpose_kernel(const float* W, bf16_t* Wt, int K, int N) {
  __shared__ float tile[32][33];
  int bn = blockIdx.x * 32, bk = blockIdx.y * 32;
  for (int i = threadIdx.y; i < 32; i += 8)
    tile[i][threadIdx.x] = W[(long long)(bk + i) * N + bn + threadIdx.x];
  __syncthreads();
  for (int i = threadIdx.y; i < 32; i += 8)
    Wt[(long long)(bn + i) * K + bk + threadIdx.x] = (bf16_t)tile[threadIdx.x][i];
}

// ---------------------------------------------------------------------------
static void launch_gemm(hipStream_t s, const GemmP& p, int M, int N, int batch) {
  if ((N & 127) == 0) {
    dim3 g(N / 128, M / BM, batch);
    gemm_wmma<128><<<g, dim3(256), 0, s>>>(p);
  } else {
    dim3 g(N / 64, M / BM, batch);
    gemm_wmma<64><<<g, dim3(256), 0, s>>>(p);
  }
}

extern "C" void kernel_launch(void* const* d_in, const int* in_sizes, int n_in,
                              void* d_out, int out_size, void* d_ws, size_t ws_size,
                              hipStream_t stream) {
  const float* q    = (const float*)d_in[0];
  const float* k    = (const float*)d_in[1];
  const float* Wq   = (const float*)d_in[2];
  const float* bq   = (const float*)d_in[3];
  const float* Wk   = (const float*)d_in[4];
  const float* bk   = (const float*)d_in[5];
  const float* Wv   = (const float*)d_in[6];
  const float* bv   = (const float*)d_in[7];
  const float* Wv0  = (const float*)d_in[8];
  const float* bv0  = (const float*)d_in[9];
  const float* Wq0  = (const float*)d_in[10];
  const float* bq0  = (const float*)d_in[11];
  const float* nq_a = (const float*)d_in[12];
  const float* nq_b = (const float*)d_in[13];
  const float* nx_a = (const float*)d_in[14];
  const float* nx_b = (const float*)d_in[15];
  const float* W1   = (const float*)d_in[16];
  const float* b1   = (const float*)d_in[17];
  const float* W2   = (const float*)d_in[18];
  const float* b2   = (const float*)d_in[19];
  const float* ns_a = (const float*)d_in[20];
  const float* ns_b = (const float*)d_in[21];

  // outputs: (x [8,1024], qq [8,512,1024], kp [8,1024,768], att_out [8,512,1024])
  float* out_x   = (float*)d_out;
  float* out_qq  = out_x  + (size_t)B_SZ * HID_;
  float* out_kp  = out_qq + (size_t)B_SZ * LQ_ * HID_;
  float* out_att = out_kp + (size_t)B_SZ * LK_ * QD_;

  // workspace carve
  char* w = (char*)d_ws; size_t off = 0;
  auto alloc = [&](size_t bytes) -> void* {
    void* p = w + off; off = (off + bytes + 255) & ~(size_t)255; return p;
  };
  bf16_t* qp_b   = (bf16_t*)alloc((size_t)B_SZ * LQ_ * QD_ * 2);
  bf16_t* kp_b   = (bf16_t*)alloc((size_t)B_SZ * LK_ * QD_ * 2);
  bf16_t* q_b    = (bf16_t*)alloc((size_t)B_SZ * LQ_ * QD_ * 2);
  bf16_t* Wtq    = (bf16_t*)alloc((size_t)QD_ * HID_ * 2);
  bf16_t* Wtk    = (bf16_t*)alloc((size_t)QD_ * HID_ * 2);
  bf16_t* Wtv    = (bf16_t*)alloc((size_t)QD_ * HID_ * 2);
  bf16_t* Wtv0   = (bf16_t*)alloc((size_t)QD_ * HID_ * 2);
  bf16_t* Wtq0   = (bf16_t*)alloc((size_t)QD_ * HID_ * 2);
  bf16_t* W1t    = (bf16_t*)alloc((size_t)HID_ * MID_ * 2);
  bf16_t* W2t    = (bf16_t*)alloc((size_t)MID_ * HID_ * 2);
  bf16_t* qh     = (bf16_t*)alloc((size_t)B_SZ * NH_ * LQ_ * DH_ * 2);
  bf16_t* kh     = (bf16_t*)alloc((size_t)B_SZ * NH_ * LK_ * DH_ * 2);
  bf16_t* v1t    = (bf16_t*)alloc((size_t)B_SZ * NH_ * DH_ * LK_ * 2);
  bf16_t* v0b    = (bf16_t*)alloc((size_t)B_SZ * NH_ * LK_ * DH_ * 2);
  float*  q0p    = (float*) alloc((size_t)B_SZ * LQ_ * HID_ * 4);
  bf16_t* q0pt   = (bf16_t*)alloc((size_t)B_SZ * NH_ * DH_ * LQ_ * 2);
  float*  scores = (float*) alloc((size_t)B_SZ * NH_ * LQ_ * LK_ * 4);
  bf16_t* att    = (bf16_t*)alloc((size_t)B_SZ * NH_ * LQ_ * LK_ * 2);
  bf16_t* attT   = (bf16_t*)alloc((size_t)B_SZ * NH_ * LK_ * LQ_ * 2);
  float*  attedr = (float*) alloc((size_t)B_SZ * LQ_ * HID_ * 4);
  float*  attedf = (float*) alloc((size_t)B_SZ * LQ_ * HID_ * 4);
  bf16_t* attedb = (bf16_t*)alloc((size_t)B_SZ * LQ_ * HID_ * 2);
  float*  xacc   = (float*) alloc((size_t)B_SZ * HID_ * 4);
  bf16_t* h1     = (bf16_t*)alloc((size_t)B_SZ * LQ_ * MID_ * 2);
  float*  h2     = (float*) alloc((size_t)B_SZ * LQ_ * HID_ * 4);
  (void)ws_size; (void)in_sizes; (void)n_in; (void)out_size;

  // --- prep: pos-enc adds + bf16 conversions ---
  addpe_kernel<<<(B_SZ * LQ_ * QD_) / 256, 256, 0, stream>>>(q, qp_b, nullptr, LQ_, QD_);
  addpe_kernel<<<(B_SZ * LK_ * QD_) / 256, 256, 0, stream>>>(k, kp_b, out_kp, LK_, QD_);
  cvt_kernel<<<(B_SZ * LQ_ * QD_) / 256, 256, 0, stream>>>(q, q_b);

  dim3 tb(32, 8);
  transpose_kernel<<<dim3(HID_ / 32, QD_ / 32), tb, 0, stream>>>(Wq,  Wtq,  QD_, HID_);
  transpose_kernel<<<dim3(HID_ / 32, QD_ / 32), tb, 0, stream>>>(Wk,  Wtk,  QD_, HID_);
  transpose_kernel<<<dim3(HID_ / 32, QD_ / 32), tb, 0, stream>>>(Wv,  Wtv,  QD_, HID_);
  transpose_kernel<<<dim3(HID_ / 32, QD_ / 32), tb, 0, stream>>>(Wv0, Wtv0, QD_, HID_);
  transpose_kernel<<<dim3(HID_ / 32, QD_ / 32), tb, 0, stream>>>(Wq0, Wtq0, QD_, HID_);
  transpose_kernel<<<dim3(MID_ / 32, HID_ / 32), tb, 0, stream>>>(W1, W1t, HID_, MID_);
  transpose_kernel<<<dim3(HID_ / 32, MID_ / 32), tb, 0, stream>>>(W2, W2t, MID_, HID_);

  GemmP p{};

  // qh[b][h][l][d]
  p = GemmP{}; p.scale = 1.f; p.bdiv = 1;
  p.A = qp_b; p.aB = (long long)LQ_ * QD_; p.lda = QD_;
  p.Bt = Wtq; p.bB = 0; p.ldbt = QD_; p.bias = bq; p.K = QD_;
  p.outB = qh; p.o_s1 = (long long)NH_ * LQ_ * DH_; p.o_sM = DH_; p.o_sH = (long long)LQ_ * DH_; p.o_sD = 1;
  launch_gemm(stream, p, LQ_, HID_, B_SZ);

  // kh[b][h][k][d]
  p = GemmP{}; p.scale = 1.f; p.bdiv = 1;
  p.A = kp_b; p.aB = (long long)LK_ * QD_; p.lda = QD_;
  p.Bt = Wtk; p.bB = 0; p.ldbt = QD_; p.bias = bk; p.K = QD_;
  p.outB = kh; p.o_s1 = (long long)NH_ * LK_ * DH_; p.o_sM = DH_; p.o_sH = (long long)LK_ * DH_; p.o_sD = 1;
  launch_gemm(stream, p, LK_, HID_, B_SZ);

  // v1t[b][h][d][k]  (transposed for att@v1)
  p = GemmP{}; p.scale = 1.f; p.bdiv = 1;
  p.A = kp_b; p.aB = (long long)LK_ * QD_; p.lda = QD_;
  p.Bt = Wtv; p.bB = 0; p.ldbt = QD_; p.bias = bv; p.K = QD_;
  p.outB = v1t; p.o_s1 = (long long)NH_ * DH_ * LK_; p.o_sM = 1; p.o_sH = (long long)DH_ * LK_; p.o_sD = LK_;
  launch_gemm(stream, p, LK_, HID_, B_SZ);

  // v0b[b][h][k][d]
  p = GemmP{}; p.scale = 1.f; p.bdiv = 1;
  p.A = kp_b; p.aB = (long long)LK_ * QD_; p.lda = QD_;
  p.Bt = Wtv0; p.bB = 0; p.ldbt = QD_; p.bias = bv0; p.K = QD_;
  p.outB = v0b; p.o_s1 = (long long)NH_ * LK_ * DH_; p.o_sM = DH_; p.o_sH = (long long)LK_ * DH_; p.o_sD = 1;
  launch_gemm(stream, p, LK_, HID_, B_SZ);

  // q0p f32 [b][l][hid]  +  q0pt bf16 [b][h][d][l]
  p = GemmP{}; p.scale = 1.f; p.bdiv = 1;
  p.A = q_b; p.aB = (long long)LQ_ * QD_; p.lda = QD_;
  p.Bt = Wtq0; p.bB = 0; p.ldbt = QD_; p.bias = bq0; p.K = QD_;
  p.outF = q0p; p.f_s1 = (long long)LQ_ * HID_; p.f_sM = HID_; p.f_sH = DH_; p.f_sD = 1;
  p.outB = q0pt; p.o_s1 = (long long)NH_ * DH_ * LQ_; p.o_sM = 1; p.o_sH = (long long)DH_ * LQ_; p.o_sD = LQ_;
  launch_gemm(stream, p, LQ_, HID_, B_SZ);

  // --- scores = qh @ kh^T / 8  (batch = B*H, Bt = kh directly) ---
  p = GemmP{}; p.scale = 0.125f; p.bdiv = 1;
  p.A = qh; p.aB = (long long)LQ_ * DH_; p.lda = DH_;
  p.Bt = kh; p.bB = (long long)LK_ * DH_; p.ldbt = DH_; p.K = DH_;
  p.outF = scores; p.f_s1 = (long long)LQ_ * LK_; p.f_sM = LK_; p.f_sH = DH_; p.f_sD = 1;
  launch_gemm(stream, p, LQ_, LK_, B_SZ * NH_);

  // att_out (relu of raw scores, mean over heads) — before softmax consumers
  attout_kernel<<<dim3(LQ_, B_SZ), 256, 0, stream>>>(scores, out_att);

  // softmax -> att (bf16) and attT (bf16)
  softmax_kernel<<<dim3(LQ_, B_SZ * NH_), 256, 0, stream>>>(scores, att, attT);

  // --- atted_raw[b][l][h*64+d] = att @ v1 ---
  p = GemmP{}; p.scale = 1.f; p.bdiv = NH_;
  p.A = att; p.aB = (long long)LQ_ * LK_; p.lda = LK_;
  p.Bt = v1t; p.bB = (long long)DH_ * LK_; p.ldbt = LK_; p.K = LK_;
  p.outF = attedr; p.f_s1 = (long long)LQ_ * HID_; p.f_s2 = DH_; p.f_sM = HID_; p.f_sH = 0; p.f_sD = 1;
  launch_gemm(stream, p, LQ_, DH_, B_SZ * NH_);

  // --- bilinear: x[b,h,d] = sum_qk (attT @ q0h)[qk,d] * v0[qk,d] ---
  zero_kernel<<<(B_SZ * HID_ + 255) / 256, 256, 0, stream>>>(xacc, B_SZ * HID_);
  p = GemmP{}; p.scale = 1.f; p.bdiv = NH_;
  p.A = attT; p.aB = (long long)LK_ * LQ_; p.lda = LQ_;
  p.Bt = q0pt; p.bB = (long long)DH_ * LQ_; p.ldbt = LQ_; p.K = LQ_;
  p.v0 = v0b; p.xacc = xacc;
  launch_gemm(stream, p, LK_, DH_, B_SZ * NH_);

  // atted = LN(q0p + atted_raw)
  ln_kernel<<<B_SZ * LQ_, 256, 0, stream>>>(q0p, attedr, nq_a, nq_b, attedf, attedb, HID_);

  // x = LN(xacc) -> output 0
  ln_kernel<<<B_SZ, 256, 0, stream>>>(xacc, nullptr, nx_a, nx_b, out_x, nullptr, HID_);

  // --- FFN: h1 = relu(atted @ W1 + b1) ; h2 = h1 @ W2 + b2 ---
  p = GemmP{}; p.scale = 1.f; p.bdiv = 1; p.relu = 1;
  p.A = attedb; p.aB = 0; p.lda = HID_;
  p.Bt = W1t; p.bB = 0; p.ldbt = HID_; p.bias = b1; p.K = HID_;
  p.outB = h1; p.o_sM = MID_; p.o_sH = DH_; p.o_sD = 1;
  launch_gemm(stream, p, B_SZ * LQ_, MID_, 1);

  p = GemmP{}; p.scale = 1.f; p.bdiv = 1;
  p.A = h1; p.aB = 0; p.lda = MID_;
  p.Bt = W2t; p.bB = 0; p.ldbt = MID_; p.bias = b2; p.K = MID_;
  p.outF = h2; p.f_sM = HID_; p.f_sH = DH_; p.f_sD = 1;
  launch_gemm(stream, p, B_SZ * LQ_, HID_, 1);

  // qq = LN(atted + ffn) -> output 1
  ln_kernel<<<B_SZ * LQ_, 256, 0, stream>>>(attedf, h2, ns_a, ns_b, out_qq, nullptr, HID_);
}
